// HiLo_4724464025685
// MI455X (gfx1250) — matrix operand order repaired and checked by reading, not compiled, not run on hardware
//
#include <hip/hip_runtime.h>

typedef unsigned short u16_t;
typedef unsigned int   u32_t;
typedef __attribute__((ext_vector_type(16))) __bf16 v16bf;
typedef __attribute__((ext_vector_type(8)))  float  v8f;

// ---------- bf16 helpers ----------
__device__ __forceinline__ float bf2f(u16_t h) {
    return __uint_as_float(((u32_t)h) << 16);
}
__device__ __forceinline__ u16_t f2bf(float f) {
    u32_t u = __float_as_uint(f);
    u += 0x7fffu + ((u >> 16) & 1u);            // round-to-nearest-even
    return (u16_t)(u >> 16);
}
__device__ __forceinline__ void unp8(uint4 v, float* f) {
    u32_t w[4] = {v.x, v.y, v.z, v.w};
#pragma unroll
    for (int i = 0; i < 4; ++i) {
        f[2 * i]     = __uint_as_float(w[i] << 16);
        f[2 * i + 1] = __uint_as_float(w[i] & 0xffff0000u);
    }
}
__device__ __forceinline__ uint4 pk8(const float* f) {
    uint4 r;
    r.x = ((u32_t)f2bf(f[1]) << 16) | f2bf(f[0]);
    r.y = ((u32_t)f2bf(f[3]) << 16) | f2bf(f[2]);
    r.z = ((u32_t)f2bf(f[5]) << 16) | f2bf(f[4]);
    r.w = ((u32_t)f2bf(f[7]) << 16) | f2bf(f[6]);
    return r;
}

union Frag16 { v16bf v; uint4 q[2]; u16_t h[16]; };

__device__ __forceinline__ v8f wmma_bf16(const Frag16& a, const Frag16& b, v8f c) {
    return __builtin_amdgcn_wmma_f32_16x16x32_bf16(false, a.v, false, b.v,
                                                   (short)0, c, false, false);
}

// window-ordered row -> natural row (B=8, N=4096, H=W=64, WS=2, 32x32 windows)
__device__ __forceinline__ int window_remap(int r) {
    int b  = r >> 12;
    int rr = r & 4095;
    int g  = rr >> 2, t = rr & 3;
    int gh = g >> 5,  gw = g & 31;
    int h  = (gh << 1) + (t >> 1);
    int w  = (gw << 1) + (t & 1);
    return (b << 12) + (h << 6) + w;
}

// ---------- fp32 -> bf16 convert (vectorized x4) ----------
__global__ __launch_bounds__(256) void cvt_bf16(const float* __restrict__ in,
                                                u16_t* __restrict__ out, int n4) {
    int i = blockIdx.x * 256 + threadIdx.x;
    if (i >= n4) return;
    float4 v = ((const float4*)in)[i];
    ushort4 r;
    r.x = f2bf(v.x); r.y = f2bf(v.y); r.z = f2bf(v.z); r.w = f2bf(v.w);
    ((ushort4*)out)[i] = r;
}

// ---------- fp32 [K,N] -> bf16 transposed [N,K] ----------
__global__ __launch_bounds__(256) void cvt_bf16_t(const float* __restrict__ in,
                                                  u16_t* __restrict__ out,
                                                  int K, int N, int total4) {
    int i = blockIdx.x * 256 + threadIdx.x;      // over N*K/4
    if (i >= total4) return;
    int kq = K >> 2;
    int k = (i % kq) * 4, n = i / kq;
    ushort4 r;
    r.x = f2bf(in[(size_t)(k + 0) * N + n]);
    r.y = f2bf(in[(size_t)(k + 1) * N + n]);
    r.z = f2bf(in[(size_t)(k + 2) * N + n]);
    r.w = f2bf(in[(size_t)(k + 3) * N + n]);
    ((ushort4*)out)[i] = r;                      // out[n*K + k .. k+3]
}

// ---------- 2x2 avg pool: x[B,64,64,512] -> xp[B,1024,512] bf16 ----------
__global__ __launch_bounds__(256) void pool_kern(const float* __restrict__ x,
                                                 u16_t* __restrict__ xp) {
    int i  = blockIdx.x * 256 + threadIdx.x;     // B*G*512 = 4194304 threads
    int c  = i & 511;
    int g  = (i >> 9) & 1023;
    int b  = i >> 19;
    int gh = g >> 5, gw = g & 31;
    int n0 = ((gh << 1) << 6) + (gw << 1);
    size_t base = ((size_t)(b << 12)) * 512;
    float s = x[base + (size_t)n0 * 512 + c] + x[base + (size_t)(n0 + 1) * 512 + c] +
              x[base + (size_t)(n0 + 64) * 512 + c] + x[base + (size_t)(n0 + 65) * 512 + c];
    xp[((size_t)(b << 10) + g) * 512 + c] = f2bf(0.25f * s);
}

// ---------- bf16 WMMA GEMM: C[M,N] = A[M,K] @ Bt[N,K]^T (+bias) ----------
// block = 256 threads = 8 waves (2x4), wave tile 32x32 (4 WMMA accums),
// block tile 64(M) x 128(N). MODE: 0=f32 out (+bias,+colOff), 1=bf16 out,
// 2=kv split (cols<256 -> C as [*,256]; cols>=256 -> C2 transposed [b,d,g])
template <int REMAP, int MODE>
__global__ __launch_bounds__(256) void gemm_wmma(
    const u16_t* __restrict__ A, const u16_t* __restrict__ Bt, void* __restrict__ C,
    u16_t* __restrict__ C2, const float* __restrict__ bias,
    int K, int lda, int ldc, int colOff, int nBlocksN) {
    int bm = blockIdx.x / nBlocksN, bn = blockIdx.x % nBlocksN;
    int wave = threadIdx.x >> 5, lane = threadIdx.x & 31;
    int wm = wave >> 2, wn = wave & 3;
    int m0 = bm * 64 + wm * 32;
    int n0 = bn * 128 + wn * 32;
    int col = lane & 15, half = lane >> 4;

    int rowA0 = m0 + col, rowA1 = m0 + 16 + col;
    if (REMAP) { rowA0 = window_remap(rowA0); rowA1 = window_remap(rowA1); }
    const u16_t* a0p = A + (size_t)rowA0 * lda + 8 * half;
    const u16_t* a1p = A + (size_t)rowA1 * lda + 8 * half;
    const u16_t* b0p = Bt + (size_t)(n0 + col) * K + 8 * half;
    const u16_t* b1p = Bt + (size_t)(n0 + 16 + col) * K + 8 * half;

    v8f c00 = {}, c01 = {}, c10 = {}, c11 = {};
    for (int k0 = 0; k0 < K; k0 += 32) {
        Frag16 a0, a1, b0, b1;
        a0.q[0] = *(const uint4*)(a0p + k0); a0.q[1] = *(const uint4*)(a0p + k0 + 16);
        a1.q[0] = *(const uint4*)(a1p + k0); a1.q[1] = *(const uint4*)(a1p + k0 + 16);
        b0.q[0] = *(const uint4*)(b0p + k0); b0.q[1] = *(const uint4*)(b0p + k0 + 16);
        b1.q[0] = *(const uint4*)(b1p + k0); b1.q[1] = *(const uint4*)(b1p + k0 + 16);
        c00 = wmma_bf16(a0, b0, c00);
        c01 = wmma_bf16(a0, b1, c01);
        c10 = wmma_bf16(a1, b0, c10);
        c11 = wmma_bf16(a1, b1, c11);
    }

    auto storeTile = [&](const v8f& cc, int mBase, int nBase) {
        int cn = nBase + col;
        float bv = (MODE == 0 && bias) ? bias[cn] : 0.0f;
#pragma unroll
        for (int r = 0; r < 8; ++r) {
            int row = mBase + r + 8 * half;
            float v = cc[r] + bv;
            if (MODE == 0) {
                ((float*)C)[(size_t)row * ldc + colOff + cn] = v;
            } else if (MODE == 1) {
                ((u16_t*)C)[(size_t)row * ldc + colOff + cn] = f2bf(v);
            } else {
                if (cn < 256)
                    ((u16_t*)C)[(size_t)row * 256 + cn] = f2bf(v);
                else  // V half, store transposed: vt[(b*256 + d)*1024 + g]
                    C2[((size_t)(row >> 10) * 256 + (cn - 256)) * 1024 + (row & 1023)] = f2bf(v);
            }
        }
    };
    storeTile(c00, m0, n0);
    storeTile(c01, m0, n0 + 16);
    storeTile(c10, m0 + 16, n0);
    storeTile(c11, m0 + 16, n0 + 16);
}

// ---------- HiFi local attention: 4-token windows, 1 thread / (b,g,t,head) ----------
__global__ __launch_bounds__(256) void hifi_attn(const u16_t* __restrict__ qkv,
                                                 u16_t* __restrict__ o_h) {
    int tid  = blockIdx.x * 256 + threadIdx.x;   // 131072 total
    int head = tid & 3;
    int t    = (tid >> 2) & 3;
    int g    = (tid >> 4) & 1023;
    int b    = tid >> 14;
    size_t wbase = ((size_t)(b << 10) + g) << 2; // first window-ordered row

    const u16_t* qp = qkv + (wbase + t) * 768 + head * 64;
    float s[4] = {0.f, 0.f, 0.f, 0.f};
#pragma unroll
    for (int ch = 0; ch < 8; ++ch) {
        float qf[8];
        unp8(*(const uint4*)(qp + ch * 8), qf);
#pragma unroll
        for (int j = 0; j < 4; ++j) {
            float kf[8];
            unp8(*(const uint4*)(qkv + (wbase + j) * 768 + 256 + head * 64 + ch * 8), kf);
#pragma unroll
            for (int d = 0; d < 8; ++d) s[j] = fmaf(qf[d], kf[d], s[j]);
        }
    }
#pragma unroll
    for (int j = 0; j < 4; ++j) s[j] *= 0.125f;  // SCALE = 64^-0.5
    float m = fmaxf(fmaxf(s[0], s[1]), fmaxf(s[2], s[3]));
    float a[4], sum = 0.f;
#pragma unroll
    for (int j = 0; j < 4; ++j) { a[j] = __expf(s[j] - m); sum += a[j]; }
    float inv = 1.0f / sum;
#pragma unroll
    for (int j = 0; j < 4; ++j) a[j] *= inv;

    int gh = g >> 5, gw = g & 31;
    int nrow = (b << 12) + (((gh << 1) + (t >> 1)) << 6) + ((gw << 1) + (t & 1));
    u16_t* op = o_h + (size_t)nrow * 256 + head * 64;
#pragma unroll
    for (int ch = 0; ch < 8; ++ch) {
        float o8[8] = {0, 0, 0, 0, 0, 0, 0, 0};
#pragma unroll
        for (int j = 0; j < 4; ++j) {
            float vf[8];
            unp8(*(const uint4*)(qkv + (wbase + j) * 768 + 512 + head * 64 + ch * 8), vf);
#pragma unroll
            for (int d = 0; d < 8; ++d) o8[d] = fmaf(a[j], vf[d], o8[d]);
        }
        *(uint4*)(op + ch * 8) = pk8(o8);
    }
}

// ---------- fused LoFi attention: per block = (b, head, 16-row Q tile) ----------
__global__ __launch_bounds__(256) void lofi_attn(const u16_t* __restrict__ q_l,
                                                 const u16_t* __restrict__ kvK,
                                                 const u16_t* __restrict__ vt,
                                                 u16_t* __restrict__ o_l) {
    __shared__ u16_t S[16 * 1024];     // scores -> probs (bf16), 32KB
    __shared__ float red[16 * 16];
    __shared__ float rowInv[16];
    __shared__ float Ob[2 * 16 * 64];  // K-split partials, 8KB

    int qt   = blockIdx.x & 255;
    int head = (blockIdx.x >> 8) & 3;
    int b    = blockIdx.x >> 10;
    int wave = threadIdx.x >> 5, lane = threadIdx.x & 31;
    int col = lane & 15, half = lane >> 4;

    size_t qrow0 = (size_t)b * 4096 + qt * 16;
    const u16_t* kvbK = kvK + ((size_t)b << 10) * 256;

    // ---- S = Q @ K^T over all 1024 pooled keys (K-dim = 64 -> 2 WMMA steps) ----
    const u16_t* aq = q_l + (qrow0 + col) * 256 + head * 64 + 8 * half;
    Frag16 a0, a1;
    a0.q[0] = *(const uint4*)(aq);       a0.q[1] = *(const uint4*)(aq + 16);
    a1.q[0] = *(const uint4*)(aq + 32);  a1.q[1] = *(const uint4*)(aq + 48);
    for (int ct = wave; ct < 64; ct += 8) {
        int g = ct * 16 + col;
        const u16_t* kp = kvbK + (size_t)g * 256 + head * 64 + 8 * half;
        Frag16 b0, b1;
        b0.q[0] = *(const uint4*)(kp);       b0.q[1] = *(const uint4*)(kp + 16);
        b1.q[0] = *(const uint4*)(kp + 32);  b1.q[1] = *(const uint4*)(kp + 48);
        v8f c = {};
        c = wmma_bf16(a0, b0, c);
        c = wmma_bf16(a1, b1, c);
#pragma unroll
        for (int r = 0; r < 8; ++r)
            S[(r + 8 * half) * 1024 + ct * 16 + col] = f2bf(c[r]);
    }
    __syncthreads();

    // ---- row softmax over 1024 (16 threads per row, 64 cols each, 8-wide LDS ops) ----
    {
        int row = threadIdx.x >> 4, seg = threadIdx.x & 15;
        u16_t* Sr = S + row * 1024 + seg * 64;
        float m = -1e30f;
#pragma unroll
        for (int c8 = 0; c8 < 8; ++c8) {
            float f[8];
            unp8(*(const uint4*)(Sr + c8 * 8), f);
#pragma unroll
            for (int d = 0; d < 8; ++d) m = fmaxf(m, f[d]);
        }
        red[row * 16 + seg] = m;
        __syncthreads();
        float rm = -1e30f;
#pragma unroll
        for (int i = 0; i < 16; ++i) rm = fmaxf(rm, red[row * 16 + i]);
        rm *= 0.125f;
        __syncthreads();
        float sum = 0.f;
#pragma unroll
        for (int c8 = 0; c8 < 8; ++c8) {
            float f[8], e[8];
            unp8(*(const uint4*)(Sr + c8 * 8), f);
#pragma unroll
            for (int d = 0; d < 8; ++d) { e[d] = __expf(f[d] * 0.125f - rm); sum += e[d]; }
            *(uint4*)(Sr + c8 * 8) = pk8(e);   // unnormalized probs; 1/rowsum folded later
        }
        red[row * 16 + seg] = sum;
        __syncthreads();
        float rs = 0.f;
#pragma unroll
        for (int i = 0; i < 16; ++i) rs += red[row * 16 + i];
        if (seg == 0) rowInv[row] = 1.0f / rs;
    }
    __syncthreads();

    // ---- O = P @ V : waves 0-3 take K[0,512), waves 4-7 take K[512,1024) ----
    {
        int n0 = (wave & 3) * 16;
        int khalf = wave >> 2;
        // V^T row for this lane's output dim: vt[(b*256 + head*64 + n0+col)][g]
        const u16_t* vrow = vt + ((size_t)b * 256 + head * 64 + n0 + col) * 1024 + 8 * half;
        v8f oc = {};
        for (int k0 = khalf * 512; k0 < khalf * 512 + 512; k0 += 32) {
            Frag16 a, bf;
            const u16_t* ap = S + col * 1024 + k0 + 8 * half;   // P row = lane&15
            a.q[0] = *(const uint4*)(ap);
            a.q[1] = *(const uint4*)(ap + 16);
            bf.q[0] = *(const uint4*)(vrow + k0);
            bf.q[1] = *(const uint4*)(vrow + k0 + 16);
            oc = wmma_bf16(a, bf, oc);
        }
#pragma unroll
        for (int r = 0; r < 8; ++r)
            Ob[khalf * 1024 + (r + 8 * half) * 64 + n0 + col] = oc[r];
    }
    __syncthreads();

    // ---- reduce K-halves, apply 1/rowsum, store bf16 ----
#pragma unroll
    for (int i = 0; i < 4; ++i) {
        int e = threadIdx.x + i * 256;
        int r = e >> 6, d = e & 63;
        float v = (Ob[e] + Ob[1024 + e]) * rowInv[r];
        o_l[(qrow0 + r) * 256 + head * 64 + d] = f2bf(v);
    }
}

extern "C" void kernel_launch(void* const* d_in, const int* in_sizes, int n_in,
                              void* d_out, int out_size, void* d_ws, size_t ws_size,
                              hipStream_t stream) {
    const float* x        = (const float*)d_in[0];
    const float* h_qkv_w  = (const float*)d_in[1];
    const float* h_proj_w = (const float*)d_in[2];
    const float* h_proj_b = (const float*)d_in[3];
    const float* l_q_w    = (const float*)d_in[4];
    const float* l_kv_w   = (const float*)d_in[5];
    const float* l_proj_w = (const float*)d_in[6];
    const float* l_proj_b = (const float*)d_in[7];
    float* out = (float*)d_out;

    // workspace layout (bf16 elements); total ~146 MB
    u16_t* ws = (u16_t*)d_ws;
    size_t off = 0;
    u16_t* xb    = ws + off; off += 16777216;   // x bf16 [32768,512]
    u16_t* xp    = ws + off; off += 4194304;    // pooled [8192,512]
    u16_t* qkvh  = ws + off; off += 25165824;   // hifi qkv [32768,768] window order
    u16_t* o_h   = ws + off; off += 8388608;    // hifi attn out [32768,256]
    u16_t* q_l   = ws + off; off += 8388608;    // lofi q [32768,256]
    u16_t* kvK   = ws + off; off += 2097152;    // lofi K [8192,256]
    u16_t* vt    = ws + off; off += 2097152;    // lofi V transposed [8*256,1024]
    u16_t* o_l   = ws + off; off += 8388608;    // lofi attn out [32768,256]
    u16_t* w_qkv = ws + off; off += 393216;     // [768,512]  (transposed)
    u16_t* w_hp  = ws + off; off += 65536;      // [256,256]
    u16_t* w_lq  = ws + off; off += 131072;     // [256,512]
    u16_t* w_lkv = ws + off; off += 262144;     // [512,512]
    u16_t* w_lp  = ws + off; off += 65536;      // [256,256]

    // x -> bf16, weights -> bf16 transposed [N,K]
    cvt_bf16<<<16777216 / 4 / 256, 256, 0, stream>>>(x, xb, 16777216 / 4);
    auto cvtT = [&](const float* src, u16_t* dst, int K, int N) {
        int t4 = (N * K) / 4;
        cvt_bf16_t<<<(t4 + 255) / 256, 256, 0, stream>>>(src, dst, K, N, t4);
    };
    cvtT(h_qkv_w, w_qkv, 512, 768);
    cvtT(h_proj_w, w_hp, 256, 256);
    cvtT(l_q_w, w_lq, 512, 256);
    cvtT(l_kv_w, w_lkv, 512, 512);
    cvtT(l_proj_w, w_lp, 256, 256);
    pool_kern<<<4194304 / 256, 256, 0, stream>>>(x, xp);

    // HiFi: qkv = window-gather(x) @ h_qkv_w   [32768,512]@[512,768]
    gemm_wmma<1, 1><<<512 * 6, 256, 0, stream>>>(xb, w_qkv, (void*)qkvh, nullptr, nullptr,
                                                 512, 512, 768, 0, 6);
    hifi_attn<<<512, 256, 0, stream>>>(qkvh, o_h);
    // HiFi proj (+bias) -> out[:, 0:256]
    gemm_wmma<0, 0><<<512 * 2, 256, 0, stream>>>(o_h, w_hp, (void*)out, nullptr, h_proj_b,
                                                 256, 256, 512, 0, 2);
    // LoFi: q = x @ l_q_w   [32768,512]@[512,256]
    gemm_wmma<0, 1><<<512 * 2, 256, 0, stream>>>(xb, w_lq, (void*)q_l, nullptr, nullptr,
                                                 512, 512, 256, 0, 2);
    // LoFi: kv = xp @ l_kv_w  [8192,512]@[512,512]; K normal, V scatter-transposed
    gemm_wmma<0, 2><<<128 * 4, 256, 0, stream>>>(xp, w_lkv, (void*)kvK, vt, nullptr,
                                                 512, 512, 256, 0, 4);
    // fused LoFi attention (flash-style, no score materialization)
    lofi_attn<<<8192, 256, 0, stream>>>(q_l, kvK, vt, o_l);
    // LoFi proj (+bias) -> out[:, 256:512]
    gemm_wmma<0, 0><<<512 * 2, 256, 0, stream>>>(o_l, w_lp, (void*)out, nullptr, l_proj_b,
                                                 256, 256, 512, 256, 2);
}
